// NTXentThetaContrastiveLoss_40707700031787
// MI455X (gfx1250) — matrix operand order repaired and checked by reading, not compiled, not run on hardware
//
#include <hip/hip_runtime.h>
#include <hip/hip_bf16.h>
#include <limits.h>

#define NN 8192
#define DD 512
#define S_SLICES 4
#define CHUNKS_TOTAL (NN / 16)                      // 512 column panels
#define CHUNKS_PER_SLICE (CHUNKS_TOTAL / S_SLICES)  // 128
#define ROWS_PER_WG 128
#define NUM_ROW_BLOCKS (NN / ROWS_PER_WG)           // 64

typedef __attribute__((ext_vector_type(16))) _Float16 v16h;
typedef __attribute__((ext_vector_type(8)))  float    v8f;
typedef __attribute__((ext_vector_type(4)))  float    f32x4;
typedef __attribute__((ext_vector_type(4)))  int      v4i;

union AV { v16h v; f32x4 q[2]; };

#if __has_builtin(__builtin_amdgcn_global_load_async_to_lds_b128)
#define HAS_ASYNC_LDS 1
#else
#define HAS_ASYNC_LDS 0
#endif

#if HAS_ASYNC_LDS
#if __has_builtin(__builtin_amdgcn_s_wait_asynccnt)
#define WAIT_ASYNC0() __builtin_amdgcn_s_wait_asynccnt(0)
#else
#define WAIT_ASYNC0() asm volatile("s_wait_asynccnt 0x0" ::: "memory")
#endif
__device__ __forceinline__ void stage_panel_async(const _Float16* src, _Float16* dst) {
    #pragma unroll
    for (int q = 0; q < 4; ++q) {
        __builtin_amdgcn_global_load_async_to_lds_b128(
            (__attribute__((address_space(1))) v4i*)(src + q * 8),
            (__attribute__((address_space(3))) v4i*)(dst + q * 8),
            0, 0);
    }
}
#endif

// ---------------------------------------------------------------- init
__global__ void init_kernel(unsigned* maxbits, float* loss, float* cnt) {
    *maxbits = 0u; *loss = 0.f; *cnt = 0.f;
}

// ------------------------------------------------- normalize rows -> fp16, theta sq
__global__ void norm_sq_kernel(const float* __restrict__ emb,
                               const float* __restrict__ th,
                               _Float16* __restrict__ e16,
                               float* __restrict__ sq) {
    __shared__ float red[256];
    const int row = blockIdx.x, tid = threadIdx.x;
    const float* er = emb + (size_t)row * DD;
    float x0 = er[tid], x1 = er[tid + 256];
    red[tid] = x0 * x0 + x1 * x1;
    __syncthreads();
    for (int o = 128; o > 0; o >>= 1) {
        if (tid < o) red[tid] += red[tid + o];
        __syncthreads();
    }
    const float inv = 1.f / fmaxf(sqrtf(red[0]), 1e-12f);
    _Float16* orow = e16 + (size_t)row * DD;
    orow[tid]       = (_Float16)(x0 * inv);
    orow[tid + 256] = (_Float16)(x1 * inv);
    if (tid == 0) {
        float a = th[row * 3 + 0], b = th[row * 3 + 1], c = th[row * 3 + 2];
        sq[row] = a * a + b * b + c * c;
    }
}

// ------------------------------------------------- global max squared distance
__global__ void distmax_kernel(const float* __restrict__ th,
                               const float* __restrict__ sq,
                               unsigned* maxbits) {
    __shared__ float red[256];
    const int i = blockIdx.x, tid = threadIdx.x;
    const float ax = th[i * 3], ay = th[i * 3 + 1], az = th[i * 3 + 2], asq = sq[i];
    float mx = 0.f;
    for (int j = tid; j < NN; j += 256) {
        float d2 = asq + sq[j]
                 - 2.f * (ax * th[j * 3] + ay * th[j * 3 + 1] + az * th[j * 3 + 2]);
        mx = fmaxf(mx, d2);
    }
    red[tid] = mx;
    __syncthreads();
    for (int o = 128; o > 0; o >>= 1) {
        if (tid < o) red[tid] = fmaxf(red[tid], red[tid + o]);
        __syncthreads();
    }
    if (tid == 0) atomicMax(maxbits, __float_as_uint(fmaxf(red[0], 0.f)));
}

// ------------------------------------------------- fused Gram(WMMA) + masked LSE
__global__ void __launch_bounds__(256, 1)
fused_kernel(const _Float16* __restrict__ e16,
             const float* __restrict__ th,
             const float* __restrict__ sq,
             const unsigned* __restrict__ maxbits,
             float* __restrict__ psum, int* __restrict__ pj,
             float* __restrict__ plog, int* __restrict__ png) {
    // double-buffered panels; 520-half pitch (1040B = 65*16B) kills bank conflicts
    __shared__ __align__(16) _Float16 Bs[2][16][520];

    const int tid  = threadIdx.x;
    const int wave = tid >> 5;
    const int lane = tid & 31;
    const int n    = lane & 15;   // column within tile (B/C/D layout)
    const int h    = lane >> 4;   // half of wave
    const int rowBase = blockIdx.x * ROWS_PER_WG;
    const int slice   = blockIdx.y;
    const int rowW    = rowBase + wave * 16;

    // staging role: thread t copies 64B of panel row brow
    const int brow = tid >> 4, bseg = tid & 15;

    // thresholds: dist/(max+1e-8) vs th  <=>  d2 vs (th*(max+1e-8))^2
    const float maxd2 = __uint_as_float(*maxbits);
    const float scale = sqrtf(fmaxf(maxd2, 0.f)) + 1e-8f;
    const float thP = 0.15f * scale, thN = 0.35f * scale;
    const float thP2 = thP * thP, thN2 = thN * thN;

    // A tile (16 rows x 512 K) in registers, WMMA A-layout:
    // lane (m = lane&15, h): k-step kk holds K = 32kk+8h..+7 and 32kk+16+8h..+7
    AV aT[16];
    {
        const _Float16* arow = e16 + (size_t)(rowW + n) * DD;  // A-layout M = lane&15
        #pragma unroll
        for (int kk = 0; kk < 16; ++kk) {
            aT[kk].q[0] = *reinterpret_cast<const f32x4*>(arow + kk * 32 + 8 * h);
            aT[kk].q[1] = *reinterpret_cast<const f32x4*>(arow + kk * 32 + 16 + 8 * h);
        }
    }

    // per-lane row metadata for C-layout rows: m = r + 8h
    float tix[8], tiy[8], tiz[8], sqi[8];
    #pragma unroll
    for (int r = 0; r < 8; ++r) {
        int gi = rowW + r + 8 * h;
        tix[r] = th[gi * 3]; tiy[r] = th[gi * 3 + 1]; tiz[r] = th[gi * 3 + 2];
        sqi[r] = sq[gi];
    }

    float lsum[8]; int firstJ[8]; float firstL[8]; int anyNeg = 0;
    #pragma unroll
    for (int r = 0; r < 8; ++r) { lsum[r] = 0.f; firstJ[r] = INT_MAX; firstL[r] = 0.f; }

    const int c0 = slice * CHUNKS_PER_SLICE;
    const int cEnd = c0 + CHUNKS_PER_SLICE;

    // ---- prologue: stage panel c0 into buffer 0
    {
        const _Float16* src = e16 + (size_t)(c0 * 16 + brow) * DD + bseg * 32;
        _Float16* dst = &Bs[0][brow][bseg * 32];
#if HAS_ASYNC_LDS
        stage_panel_async(src, dst);
        WAIT_ASYNC0();
#else
        const f32x4* s4 = reinterpret_cast<const f32x4*>(src);
        f32x4* d4 = reinterpret_cast<f32x4*>(dst);
        d4[0] = s4[0]; d4[1] = s4[1]; d4[2] = s4[2]; d4[3] = s4[3];
#endif
    }
    __syncthreads();

    for (int ch = c0; ch < cEnd; ++ch) {
        const int cur = (ch - c0) & 1;
        const int nxt = cur ^ 1;
        const bool hasNext = (ch + 1 < cEnd);

#if !HAS_ASYNC_LDS
        f32x4 rg0, rg1, rg2, rg3;   // register staging (fallback path)
#endif
        if (hasNext) {
            const _Float16* src = e16 + (size_t)((ch + 1) * 16 + brow) * DD + bseg * 32;
#if HAS_ASYNC_LDS
            stage_panel_async(src, &Bs[nxt][brow][bseg * 32]);   // overlaps with WMMAs below
#else
            const f32x4* s4 = reinterpret_cast<const f32x4*>(src);
            rg0 = s4[0]; rg1 = s4[1]; rg2 = s4[2]; rg3 = s4[3];  // loads issued early
#endif
        }

        // ---- 16 WMMAs over K = 512 from the current LDS panel
        v8f cacc = {};
        #pragma unroll
        for (int kk = 0; kk < 16; ++kk) {
            AV b;  // B-layout: lane (n, h) holds K = 32kk+8h..+7 and 32kk+16+8h..+7, column n
            b.q[0] = *reinterpret_cast<const f32x4*>(&Bs[cur][n][kk * 32 + 8 * h]);
            b.q[1] = *reinterpret_cast<const f32x4*>(&Bs[cur][n][kk * 32 + 16 + 8 * h]);
            cacc = __builtin_amdgcn_wmma_f32_16x16x32_f16(
                false, aT[kk].v, false, b.v, (short)0, cacc, false, false);
        }

        // ---- masked online reduction for this 16x16 tile
        const int jBase = ch * 16;
        const int jcol = jBase + n;
        const float tjx = th[jcol * 3], tjy = th[jcol * 3 + 1], tjz = th[jcol * 3 + 2];
        const float sqj = sq[jcol];
        #pragma unroll
        for (int r = 0; r < 8; ++r) {
            float s = fminf(fmaxf(cacc[r] * 10.f, -30.f), 30.f);  // /TEMP then clip
            float d2 = sqi[r] + sqj
                     - 2.f * (tix[r] * tjx + tiy[r] * tjy + tiz[r] * tjz);
            int gi = rowW + r + 8 * h;
            bool posm = (d2 < thP2) && (gi != jcol);
            bool negm = (d2 > thN2);
            // fixed-max LSE: sim <= 30 always; exp(s-30) in [e^-60, 1] stays normal f32
            lsum[r] += (posm || negm) ? __expf(s - 30.f) : 0.f;
            if (posm && jcol < firstJ[r]) { firstJ[r] = jcol; firstL[r] = s; }
            anyNeg |= negm ? (1 << r) : 0;
        }

        if (hasNext) {
#if HAS_ASYNC_LDS
            WAIT_ASYNC0();                 // this wave's async writes to Bs[nxt] done
#else
            f32x4* d4 = reinterpret_cast<f32x4*>(&Bs[nxt][brow][bseg * 32]);
            d4[0] = rg0; d4[1] = rg1; d4[2] = rg2; d4[3] = rg3;
#endif
        }
        __syncthreads();                   // one barrier per chunk (double buffered)
    }

    // reduce across the 16 lanes of each half (rows r+8h live in lanes 16h..16h+15)
    #pragma unroll
    for (int r = 0; r < 8; ++r) {
        float s = lsum[r]; int fj = firstJ[r]; float fl = firstL[r];
        int ng = (anyNeg >> r) & 1;
        #pragma unroll
        for (int off = 1; off < 16; off <<= 1) {
            s += __shfl_xor(s, off, 32);
            int   fj2 = __shfl_xor(fj, off, 32);
            float fl2 = __shfl_xor(fl, off, 32);
            if (fj2 < fj) { fj = fj2; fl = fl2; }
            ng |= __shfl_xor(ng, off, 32);
        }
        if (n == 0) {
            int row = rowW + r + 8 * h;
            size_t idx = (size_t)row * S_SLICES + slice;
            psum[idx] = s; pj[idx] = fj; plog[idx] = fl; png[idx] = ng;
        }
    }
}

// ------------------------------------------------- merge slices, accumulate loss
__global__ void merge_kernel(const float* __restrict__ psum, const int* __restrict__ pj,
                             const float* __restrict__ plog, const int* __restrict__ png,
                             float* loss, float* cnt) {
    const int row = blockIdx.x * blockDim.x + threadIdx.x;
    if (row >= NN) return;
    float tot = 0.f; int fj = INT_MAX; float fl = 0.f; int ng = 0;
    for (int s = 0; s < S_SLICES; ++s) {
        size_t idx = (size_t)row * S_SLICES + s;
        tot += psum[idx];                 // fixed-max LSE partial sums add directly
        int j2 = pj[idx];
        if (j2 < fj) { fj = j2; fl = plog[idx]; }
        ng |= png[idx];
    }
    const bool valid = (fj != INT_MAX) && (ng != 0);
    const float per = valid ? (30.f + logf(tot) - fl) : 0.f;
    atomicAdd(loss, per);
    atomicAdd(cnt, valid ? 1.f : 0.f);
}

__global__ void finalize_kernel(const float* loss, const float* cnt, float* out) {
    out[0] = (cnt[0] > 0.f) ? (loss[0] / fmaxf(cnt[0], 1.f)) : 0.f;
}

// ---------------------------------------------------------------- launch
extern "C" void kernel_launch(void* const* d_in, const int* in_sizes, int n_in,
                              void* d_out, int out_size, void* d_ws, size_t ws_size,
                              hipStream_t stream) {
    const float* emb = (const float*)d_in[0];   // [N, D] f32
    const float* th  = (const float*)d_in[1];   // [N, 3] f32
    float* out = (float*)d_out;

    char* ws = (char*)d_ws;
    size_t off = 0;
    _Float16* e16  = (_Float16*)(ws + off); off += (size_t)NN * DD * sizeof(_Float16);
    float*    sq   = (float*)(ws + off);    off += (size_t)NN * sizeof(float);
    float*    psum = (float*)(ws + off);    off += (size_t)NN * S_SLICES * sizeof(float);
    int*      pj   = (int*)(ws + off);      off += (size_t)NN * S_SLICES * sizeof(int);
    float*    plog = (float*)(ws + off);    off += (size_t)NN * S_SLICES * sizeof(float);
    int*      png  = (int*)(ws + off);      off += (size_t)NN * S_SLICES * sizeof(int);
    unsigned* maxbits = (unsigned*)(ws + off); off += sizeof(unsigned);
    float*    loss = (float*)(ws + off);    off += sizeof(float);
    float*    cnt  = (float*)(ws + off);    off += sizeof(float);

    init_kernel<<<1, 1, 0, stream>>>(maxbits, loss, cnt);
    norm_sq_kernel<<<NN, 256, 0, stream>>>(emb, th, e16, sq);
    distmax_kernel<<<NN, 256, 0, stream>>>(th, sq, maxbits);
    dim3 grid(NUM_ROW_BLOCKS, S_SLICES);
    fused_kernel<<<grid, 256, 0, stream>>>(e16, th, sq, maxbits, psum, pj, plog, png);
    merge_kernel<<<NN / 256, 256, 0, stream>>>(psum, pj, plog, png, loss, cnt);
    finalize_kernel<<<1, 1, 0, stream>>>(loss, cnt, out);
}
